// SpatialTemporalAttention_89850715832970
// MI455X (gfx1250) — compile-verified, matrix operand order checked by reading, not run on hardware
//
#include <hip/hip_runtime.h>
#include <hip/hip_bf16.h>

typedef __attribute__((ext_vector_type(16))) _Float16 v16h;
typedef __attribute__((ext_vector_type(8)))  float    v8f;
typedef __attribute__((ext_vector_type(4)))  unsigned int u32x4;
typedef __attribute__((ext_vector_type(8)))  int      i32x8;
typedef __attribute__((ext_vector_type(4)))  int      i32x4;

#define WMMA_F16(a, b, c) \
  __builtin_amdgcn_wmma_f32_16x16x32_f16(false, (a), false, (b), (short)0, (c), false, false)

static constexpr int Bb = 2, Cc = 256, Tt = 24, HWn = 2304, Ff = 48, NH = 8;
static constexpr float SCALE = 0.17677669529663687f;  // 32^-0.5

// ---------------- workspace layout (bytes) ----------------
static constexpr size_t QS_BYTES   = (size_t)Ff * Cc * HWn * 2;           // f16
static constexpr size_t H1T_BYTES  = (size_t)Bb * HWn * Tt * Cc * 2;      // f16
static constexpr size_t M_BYTES    = (size_t)Ff * Cc * Cc * 2;            // f16
static constexpr size_t CTX_BYTES  = (size_t)Ff * NH * 32 * 32 * 4;       // f32
static constexpr size_t KSUM_BYTES = (size_t)Ff * Cc * 4;                 // f32
static constexpr size_t QS_OFF   = 0;
static constexpr size_t H1T_OFF  = QS_OFF + QS_BYTES;
static constexpr size_t M_OFF    = H1T_OFF + H1T_BYTES;
static constexpr size_t CTX_OFF  = M_OFF + M_BYTES;
static constexpr size_t KSUM_OFF = CTX_OFF + CTX_BYTES;

// ---------------- WMMA fragment helpers (wave32) ----------------
__device__ inline v16h frag_a_lds(const _Float16* p, int stride) {
  const int lane = threadIdx.x & 31;
  const _Float16* r = p + (lane & 15) * stride;
  const int kq = (lane >> 4) << 3;
  v16h a;
#pragma unroll
  for (int i = 0; i < 8; ++i) { a[i] = r[kq + i]; a[8 + i] = r[16 + kq + i]; }
  return a;
}
__device__ inline v16h frag_a_g32(const float* p, int stride) {
  const int lane = threadIdx.x & 31;
  const float* r = p + (size_t)(lane & 15) * stride;
  const int kq = (lane >> 4) << 3;
  v16h a;
#pragma unroll
  for (int i = 0; i < 8; ++i) { a[i] = (_Float16)r[kq + i]; a[8 + i] = (_Float16)r[16 + kq + i]; }
  return a;
}
// B (KxN) from an n-major (transposed) source: element (k,n) at p[n*stride + k]
__device__ inline v16h frag_bt_lds(const _Float16* p, int stride) {
  const int lane = threadIdx.x & 31;
  const _Float16* r = p + (lane & 15) * stride;
  const int kq = (lane >> 4) << 3;
  v16h b;
#pragma unroll
  for (int i = 0; i < 8; ++i) { b[i] = r[kq + i]; b[8 + i] = r[16 + kq + i]; }
  return b;
}
__device__ inline void c_store_f32(float* p, int stride, const v8f& c) {
  const int lane = threadIdx.x & 31;
  const int n = lane & 15, ro = (lane >> 4) << 3;
#pragma unroll
  for (int r = 0; r < 8; ++r) p[(ro + r) * stride + n] = c[r];
}
__device__ inline void c_store_f16s(_Float16* p, int stride, const v8f& c, float s) {
  const int lane = threadIdx.x & 31;
  const int n = lane & 15, ro = (lane >> 4) << 3;
#pragma unroll
  for (int r = 0; r < 8; ++r) p[(ro + r) * stride + n] = (_Float16)(c[r] * s);
}
__device__ inline void c_store_f16_T(_Float16* p, int stride, const v8f& c) {
  const int lane = threadIdx.x & 31;
  const int n = lane & 15, ro = (lane >> 4) << 3;
#pragma unroll
  for (int r = 0; r < 8; ++r) p[n * stride + ro + r] = (_Float16)c[r];
}

// ---------------- TDM: 2D tensor load global->LDS (D# per CDNA5 ISA) ----------
// data_size = 4 bytes (code 2). Dims/strides in dword units. Optional per-row LDS padding.
__device__ inline void tdm_load_2d(unsigned lds_off, const void* gptr,
                                   unsigned tensor_d0, unsigned tensor_d1,
                                   unsigned tile_d0, unsigned tile_d1,
                                   unsigned d0_stride,
                                   unsigned pad_interval_code, unsigned pad_amount_code,
                                   bool pad_en) {
  unsigned long long ga = (unsigned long long)(size_t)gptr;
  u32x4 g0;
  g0[0] = 1u;                                        // count=1 valid user descriptor
  g0[1] = lds_off;                                   // lds_addr
  g0[2] = (unsigned)(ga & 0xFFFFFFFFu);              // global_addr[31:0]
  g0[3] = (unsigned)((ga >> 32) & 0x01FFFFFFu) | (2u << 30);  // addr[56:32] | type=2
  i32x8 g1;
  g1[0] = (int)((2u << 16) |                         // data_size = 4B
                ((pad_en ? 1u : 0u) << 20) |
                (pad_interval_code << 22) | (pad_amount_code << 25));
  g1[1] = (int)((tensor_d0 & 0xFFFFu) << 16);        // tensor_dim0[15:0] @ bits 63:48
  g1[2] = (int)((tensor_d0 >> 16) | ((tensor_d1 & 0xFFFFu) << 16));
  g1[3] = (int)((tensor_d1 >> 16) | ((tile_d0 & 0xFFFFu) << 16));
  g1[4] = (int)(tile_d1 & 0xFFFFu);                  // tile_dim1; tile_dim2=0 (unused)
  g1[5] = (int)d0_stride;                            // tensor_dim0_stride[31:0]
  g1[6] = 0;
  g1[7] = 0;
  i32x4 z4 = {0, 0, 0, 0};                           // groups 2/3: zero => dims 3..5 unused
  i32x8 z8 = {0, 0, 0, 0, 0, 0, 0, 0};
  __builtin_amdgcn_tensor_load_to_lds(g0, g1, z4, z4, z8, 0);
}
__device__ inline bool wave0_scalar() {
  return __builtin_amdgcn_readfirstlane((int)threadIdx.x) < 32;  // uniform (SGPR) branch
}

// ================= Kernel 1: spatial qkv + softmaxes + partial context =================
static constexpr int BX_S = 260, AS_S = 36, KE_S = 36, V_S = 36;
__global__ __launch_bounds__(256)
void k1_spatial(const float* __restrict__ x, const float* __restrict__ Wqkv,
                _Float16* __restrict__ QS, float* __restrict__ Ksum, float* __restrict__ ctxU) {
  const int f = blockIdx.y, bi = f / Tt, ti = f % Tt;
  const int p0 = blockIdx.x * 32;
  const int tid = threadIdx.x, w = tid >> 5;

  __shared__ _Float16 BxT[32 * BX_S];    // x tile transposed: [p][c]
  __shared__ _Float16 As[256 * AS_S];    // weight chunk: [row][k]
  __shared__ float    Cs[256 * 32];      // chunk result: [row][p]
  __shared__ _Float16 KE[256 * KE_S];    // exp(k): [d][p]
  __shared__ _Float16 Vf[256 * V_S];     // v: [e][p]

  for (int idx = tid; idx < 256 * 32; idx += 256) {
    int c = idx >> 5, p = idx & 31;
    BxT[p * BX_S + c] = (_Float16)x[(((size_t)bi * Cc + c) * Tt + ti) * HWn + p0 + p];
  }
  __syncthreads();

  for (int chunk = 0; chunk < 3; ++chunk) {       // 0:q 1:k 2:v
    const int rowBase = chunk * 256;
    v8f acc[4] = {};
    for (int k0 = 0; k0 < 256; k0 += 32) {
      __syncthreads();
      for (int idx = tid; idx < 256 * 32; idx += 256) {
        int r = idx >> 5, kk = idx & 31;
        As[r * AS_S + kk] = (_Float16)Wqkv[(size_t)(rowBase + r) * 256 + k0 + kk];
      }
      if (k0 + 32 < 256)
        __builtin_prefetch(&Wqkv[(size_t)(rowBase + (tid >> 5) * 32) * 256 + k0 + 32], 0, 0);
      __syncthreads();
#pragma unroll
      for (int i = 0; i < 4; ++i) {
        int rt = 2 * w + (i >> 1), ct = i & 1;
        v16h a = frag_a_lds(&As[rt * 16 * AS_S], AS_S);
        v16h b = frag_bt_lds(&BxT[(ct * 16) * BX_S + k0], BX_S);
        acc[i] = WMMA_F16(a, b, acc[i]);
      }
    }
    __syncthreads();
#pragma unroll
    for (int i = 0; i < 4; ++i) {
      int rt = 2 * w + (i >> 1), ct = i & 1;
      c_store_f32(&Cs[rt * 16 * 32 + ct * 16], 32, acc[i]);
    }
    __syncthreads();

    if (chunk == 0) {  // q: softmax over d per pixel, * scale
      int h = tid >> 5, col = tid & 31;
      float m = -1e30f;
      for (int d = 0; d < 32; ++d) m = fmaxf(m, Cs[(h * 32 + d) * 32 + col]);
      float s = 0.f;
      for (int d = 0; d < 32; ++d) s += expf(Cs[(h * 32 + d) * 32 + col] - m);
      float inv = SCALE / s;
      for (int d = 0; d < 32; ++d)
        QS[((size_t)f * Cc + h * 32 + d) * HWn + p0 + col] =
            (_Float16)(expf(Cs[(h * 32 + d) * 32 + col] - m) * inv);
    } else if (chunk == 1) {  // k: exp + partial row sums
      float s = 0.f;
      for (int p = 0; p < 32; ++p) {
        float e = expf(Cs[tid * 32 + p]);
        KE[tid * KE_S + p] = (_Float16)e;
        s += e;
      }
      atomicAdd(&Ksum[f * Cc + tid], s);
    } else {  // v: per-head partial context = KE(32xP) x V^T(Px32)
      for (int idx = tid; idx < 256 * 32; idx += 256) {
        int r = idx >> 5, p = idx & 31;
        Vf[r * V_S + p] = (_Float16)Cs[idx];
      }
      __syncthreads();
      const int lane = tid & 31, nl = lane & 15, ro = (lane >> 4) << 3;
#pragma unroll
      for (int i = 0; i < 4; ++i) {
        int tidx = w * 4 + i;
        int hh = tidx >> 2, mt = (tidx >> 1) & 1, nt = tidx & 1;
        v16h a = frag_a_lds(&KE[(hh * 32 + mt * 16) * KE_S], KE_S);
        v16h b = frag_bt_lds(&Vf[(hh * 32 + nt * 16) * V_S], V_S);
        v8f cc = {};
        cc = WMMA_F16(a, b, cc);
#pragma unroll
        for (int r = 0; r < 8; ++r)
          atomicAdd(&ctxU[(((size_t)f * NH + hh) * 32 + mt * 16 + ro + r) * 32 + nt * 16 + nl],
                    cc[r]);
      }
    }
    __syncthreads();
  }
}

// ================= Kernel 2: normalize context, fold into out-proj =================
static constexpr int CT_S = 36;
__global__ __launch_bounds__(256)
void k2_context(const float* __restrict__ Wout, const float* __restrict__ ctxU,
                const float* __restrict__ Ksum, _Float16* __restrict__ M) {
  const int f = blockIdx.x, tid = threadIdx.x, w = tid >> 5;
  __shared__ _Float16 ctxN[NH * 32 * CT_S];  // [h][d][e]  (n-major for B: n=d, k=e)
  for (int idx = tid; idx < NH * 32 * 32; idx += 256) {
    int h = idx >> 10, d = (idx >> 5) & 31, e = idx & 31;
    float v = ctxU[(((size_t)f * NH + h) * 32 + d) * 32 + e] / Ksum[f * Cc + h * 32 + d];
    ctxN[(h * 32 + d) * CT_S + e] = (_Float16)v;
  }
  __syncthreads();
  for (int i = 0; i < 32; ++i) {
    int tile = w * 32 + i;
    int rt = tile >> 4, ctile = tile & 15, h = ctile >> 1, dh = ctile & 1;
    v16h a = frag_a_g32(&Wout[(size_t)(rt * 16) * 256 + h * 32], 256);
    v16h b = frag_bt_lds(&ctxN[(h * 32 + dh * 16) * CT_S], CT_S);
    v8f cc = {};
    cc = WMMA_F16(a, b, cc);
    c_store_f16s(&M[((size_t)f * 256 + rt * 16) * 256 + ctile * 16], 256, cc, 1.0f);
  }
}

// ================= Kernel 3: h1 = M * QS + bias =================
static constexpr int A3_S = 36, B3_S = 36;
__global__ __launch_bounds__(128)
void k3_proj_spatial(const _Float16* __restrict__ M, const _Float16* __restrict__ QS,
                     const float* __restrict__ bias, _Float16* __restrict__ h1t) {
  const int f = blockIdx.z, bi = f / Tt, ti = f % Tt;
  const int r0 = blockIdx.y * 32, p0 = blockIdx.x * 64;
  const int tid = threadIdx.x, w = tid >> 5;
  __shared__ _Float16 As[32 * A3_S];     // [row][k] via TDM (pad 16->18 dwords/row)
  __shared__ _Float16 BsT[64 * B3_S];    // QS tile transposed: [p][k]
  const unsigned as_off = (unsigned)(size_t)&As[0];
  v8f acc[2] = {};
  for (int k0 = 0; k0 < 256; k0 += 32) {
    __syncthreads();
    // TDM: 32 rows x 16 dwords from M (row stride 128 dwords), pad_interval=16(code3), amount=2(code1)
    if (wave0_scalar())
      tdm_load_2d(as_off, &M[((size_t)f * 256 + r0) * 256 + k0],
                  /*td0=*/16, /*td1=*/32, /*tile0=*/16, /*tile1=*/32,
                  /*stride0=*/128, /*padi=*/3, /*pada=*/1, true);
    for (int idx = tid; idx < 32 * 64; idx += 128) {
      int k = idx >> 6, p = idx & 63;
      BsT[p * B3_S + k] = QS[((size_t)f * Cc + k0 + k) * HWn + p0 + p];
    }
    if (k0 + 32 < 256)
      __builtin_prefetch(&M[((size_t)f * 256 + r0 + (tid & 31)) * 256 + k0 + 32], 0, 0);
    __builtin_amdgcn_s_wait_tensorcnt(0);
    __syncthreads();
    v16h a = frag_a_lds(&As[(w >> 1) * 16 * A3_S], A3_S);
#pragma unroll
    for (int i = 0; i < 2; ++i) {
      int ct = (w & 1) * 2 + i;
      v16h b = frag_bt_lds(&BsT[(ct * 16) * B3_S], B3_S);
      acc[i] = WMMA_F16(a, b, acc[i]);
    }
  }
  const int lane = tid & 31, nl = lane & 15, ro = (lane >> 4) << 3;
  const int rt = w >> 1;
#pragma unroll
  for (int i = 0; i < 2; ++i) {
    int ct = (w & 1) * 2 + i;
    int pix = p0 + ct * 16 + nl;
#pragma unroll
    for (int r = 0; r < 8; ++r) {
      int ch = r0 + rt * 16 + ro + r;
      float v = acc[i][r] + bias[ch];
      h1t[(((size_t)bi * HWn + pix) * Tt + ti) * Cc + ch] = (_Float16)v;
    }
  }
}

// ================= Kernel 4: fused temporal attention + residual =================
static constexpr int XL_S = 260, QK_S = 516, BW_S = 36, VT_S = 36, AT_S = 36;
__global__ __launch_bounds__(256)
void k4_temporal(const float* __restrict__ x, const _Float16* __restrict__ h1t,
                 const float* __restrict__ Wqkv, const float* __restrict__ Wout,
                 float* __restrict__ out) {
  const int bi = blockIdx.y, pix = blockIdx.x;
  const int tid = threadIdx.x, w = tid >> 5, lane = tid & 31;

  __shared__ _Float16 Xl[32 * XL_S];     // h1 slab [t][c], TDM-padded; later reused for o
  __shared__ _Float16 Qkv[32 * QK_S];    // [t][q(256)|k(256)]
  __shared__ _Float16 BwT[256 * BW_S];   // weight staging, [n][k]
  __shared__ _Float16 Vt4[256 * VT_S];   // v transposed: [e][t]
  __shared__ float    Sim[8][32 * 32];
  __shared__ _Float16 Attn[8][32 * AT_S];

  // TDM: 24 rows x 128 dwords (contiguous) from h1t; tile 32 rows -> rows 24..31 OOB = zero;
  // pad_interval=128 dwords (code 6), pad_amount=2 dwords (code 1) => LDS row stride 260 halfs.
  if (wave0_scalar())
    tdm_load_2d((unsigned)(size_t)&Xl[0], &h1t[(((size_t)bi * HWn + pix) * Tt) * Cc],
                /*td0=*/128, /*td1=*/Tt, /*tile0=*/128, /*tile1=*/32,
                /*stride0=*/128, /*padi=*/6, /*pada=*/1, true);
  __builtin_amdgcn_s_wait_tensorcnt(0);
  __syncthreads();

  // ---- qkv = Xl (32x256) @ Wqkv^T (256x768), in 6 column chunks of 128 ----
  for (int chunk = 0; chunk < 6; ++chunk) {
    const int colBase = chunk * 128;
    v8f acc[2] = {};
    for (int k0 = 0; k0 < 256; k0 += 32) {
      __syncthreads();
      for (int idx = tid; idx < 32 * 128; idx += 256) {
        int n = idx >> 5, k = idx & 31;
        BwT[n * BW_S + k] = (_Float16)Wqkv[(size_t)(colBase + n) * 256 + k0 + k];
      }
      if (k0 + 32 < 256)
        __builtin_prefetch(&Wqkv[(size_t)(colBase + (tid >> 1)) * 256 + k0 + 32], 0, 0);
      __syncthreads();
#pragma unroll
      for (int i = 0; i < 2; ++i) {
        int tile = w * 2 + i, rt = tile >> 3, ct = tile & 7;
        v16h a = frag_a_lds(&Xl[rt * 16 * XL_S + k0], XL_S);
        v16h b = frag_bt_lds(&BwT[(ct * 16) * BW_S], BW_S);
        acc[i] = WMMA_F16(a, b, acc[i]);
      }
    }
    float s = (chunk < 2) ? SCALE : 1.0f;
    __syncthreads();
#pragma unroll
    for (int i = 0; i < 2; ++i) {
      int tile = w * 2 + i, rt = tile >> 3, ct = tile & 7;
      if (chunk < 4) {
        c_store_f16s(&Qkv[rt * 16 * QK_S + colBase + ct * 16], QK_S, acc[i], s);
      } else {
        c_store_f16_T(&Vt4[(colBase - 512 + ct * 16) * VT_S + rt * 16], VT_S, acc[i]);
      }
    }
  }
  __syncthreads();

  // ---- sim = q @ k^T per head (wave w -> head w) ----
  {
    const int h = w;
    v8f sacc[4] = {};
#pragma unroll
    for (int mt = 0; mt < 2; ++mt)
#pragma unroll
      for (int nt = 0; nt < 2; ++nt) {
        v16h a = frag_a_lds(&Qkv[mt * 16 * QK_S + h * 32], QK_S);
        v16h b = frag_bt_lds(&Qkv[nt * 16 * QK_S + 256 + h * 32], QK_S);
        sacc[mt * 2 + nt] = WMMA_F16(a, b, sacc[mt * 2 + nt]);
      }
#pragma unroll
    for (int i = 0; i < 4; ++i)
      c_store_f32(&Sim[w][(i >> 1) * 16 * 32 + (i & 1) * 16], 32, sacc[i]);
  }
  __syncthreads();

  // ---- masked softmax over 24 valid frames (lane = row) ----
  {
    const int row = lane;
    float m = -1e30f;
    for (int j = 0; j < Tt; ++j) m = fmaxf(m, Sim[w][row * 32 + j]);
    float s = 0.f;
    for (int j = 0; j < Tt; ++j) s += expf(Sim[w][row * 32 + j] - m);
    float inv = 1.0f / s;
    for (int j = 0; j < 32; ++j)
      Attn[w][row * AT_S + j] =
          (j < Tt) ? (_Float16)(expf(Sim[w][row * 32 + j] - m) * inv) : (_Float16)0.f;
  }
  __syncthreads();

  // ---- o = attn @ v -> Xl (h1 slab dead) ----
  {
    const int h = w;
#pragma unroll
    for (int i = 0; i < 4; ++i) {
      int mt = i >> 1, nt = i & 1;
      v16h a = frag_a_lds(&Attn[w][mt * 16 * AT_S], AT_S);
      v16h b = frag_bt_lds(&Vt4[(h * 32 + nt * 16) * VT_S], VT_S);
      v8f oacc = {};
      oacc = WMMA_F16(a, b, oacc);
      c_store_f16s(&Xl[mt * 16 * XL_S + h * 32 + nt * 16], XL_S, oacc, 1.0f);
    }
  }
  __syncthreads();

  // ---- h2 = o @ Wout^T, residual add with x ----
  {
    v8f pacc[4] = {};
    for (int k0 = 0; k0 < 256; k0 += 32) {
      __syncthreads();
      for (int idx = tid; idx < 32 * 256; idx += 256) {
        int n = idx >> 5, k = idx & 31;
        BwT[n * BW_S + k] = (_Float16)Wout[(size_t)n * 256 + k0 + k];
      }
      if (k0 + 32 < 256)
        __builtin_prefetch(&Wout[(size_t)tid * 256 + k0 + 32], 0, 0);
      __syncthreads();
#pragma unroll
      for (int i = 0; i < 4; ++i) {
        int tile = w * 4 + i, rt = tile >> 4, ct = tile & 15;
        v16h a = frag_a_lds(&Xl[rt * 16 * XL_S + k0], XL_S);
        v16h b = frag_bt_lds(&BwT[(ct * 16) * BW_S], BW_S);
        pacc[i] = WMMA_F16(a, b, pacc[i]);
      }
    }
    const int nl = lane & 15, ro = (lane >> 4) << 3;
#pragma unroll
    for (int i = 0; i < 4; ++i) {
      int tile = w * 4 + i, rt = tile >> 4, ct = tile & 15;
#pragma unroll
      for (int r = 0; r < 8; ++r) {
        int t = rt * 16 + ro + r;
        if (t < Tt) {
          int ch = ct * 16 + nl;
          size_t oidx = (((size_t)bi * Cc + ch) * Tt + t) * HWn + pix;
          out[oidx] = x[oidx] + pacc[i][r];
        }
      }
    }
  }
}

// ================= host =================
extern "C" void kernel_launch(void* const* d_in, const int* in_sizes, int n_in,
                              void* d_out, int out_size, void* d_ws, size_t ws_size,
                              hipStream_t stream) {
  (void)in_sizes; (void)n_in; (void)out_size; (void)ws_size;
  const float* x      = (const float*)d_in[0];
  const float* saqkvw = (const float*)d_in[1];
  const float* saoutw = (const float*)d_in[2];
  const float* saoutb = (const float*)d_in[3];
  const float* taqkvw = (const float*)d_in[4];
  const float* taoutw = (const float*)d_in[5];
  float* out = (float*)d_out;

  char* ws = (char*)d_ws;
  _Float16* QS  = (_Float16*)(ws + QS_OFF);
  _Float16* H1T = (_Float16*)(ws + H1T_OFF);
  _Float16* Mm  = (_Float16*)(ws + M_OFF);
  float* ctxU   = (float*)(ws + CTX_OFF);
  float* Ksum   = (float*)(ws + KSUM_OFF);

  (void)hipMemsetAsync(ws + CTX_OFF, 0, CTX_BYTES + KSUM_BYTES, stream);

  k1_spatial<<<dim3(HWn / 32, Ff), 256, 0, stream>>>(x, saqkvw, QS, Ksum, ctxU);
  k2_context<<<dim3(Ff), 256, 0, stream>>>(saoutw, ctxU, Ksum, Mm);
  k3_proj_spatial<<<dim3(HWn / 64, Cc / 32, Ff), 128, 0, stream>>>(Mm, QS, saoutb, H1T);
  k4_temporal<<<dim3(HWn, Bb), 256, 0, stream>>>(x, H1T, taqkvw, taoutw, out);
}
